// PSAttn_12214886990497
// MI455X (gfx1250) — matrix operand order repaired and checked by reading, not compiled, not run on hardware
//
#include <hip/hip_runtime.h>
#include <hip/hip_bf16.h>
#include <math.h>

typedef __attribute__((ext_vector_type(16))) _Float16 v16h;
typedef __attribute__((ext_vector_type(8)))  float    v8f;

static constexpr int Bn   = 2;
static constexpr int CDIM = 256;
static constexpr int NH   = 8;
static constexpr int HD   = 32;
static constexpr int Npix = 4096;   // 64*64
static constexpr int NuPix= 1024;   // 32*32
static constexpr float SCALE_QK = 0.17677669529663687f; // 32^-0.5

__device__ __forceinline__ v8f wmma_f16(v16h a, v16h b, v8f c) {
  return __builtin_amdgcn_wmma_f32_16x16x32_f16(false, a, false, b, (short)0, c, false, false);
}

__device__ __forceinline__ v8f vzero8() {
  v8f z;
#pragma unroll
  for (int i = 0; i < 8; ++i) z[i] = 0.f;
  return z;
}

// ---------------------------------------------------------------------------
// Transpose-pack: planar f32 [b][C][P] -> pixel-major f16 [b][P][C]
// ---------------------------------------------------------------------------
__global__ void psattn_trpack_kernel(const float* __restrict__ src, _Float16* __restrict__ dst,
                                     int C, int P, size_t total) {
  size_t i = (size_t)blockIdx.x * blockDim.x + threadIdx.x;
  if (i >= total) return;
  int p = (int)(i % P);
  size_t rem = i / P;
  int o = (int)(rem % C);
  int b = (int)(rem / C);
  dst[((size_t)b * P + p) * C + o] = (_Float16)src[i];
}

// ---------------------------------------------------------------------------
// Straight convert f32 -> f16 with scale (contiguous)
// ---------------------------------------------------------------------------
__global__ void psattn_cvt16_kernel(const float* __restrict__ src, _Float16* __restrict__ dst,
                                    float scale, size_t total) {
  size_t i = (size_t)blockIdx.x * blockDim.x + threadIdx.x;
  if (i >= total) return;
  dst[i] = (_Float16)(src[i] * scale);
}

// ---------------------------------------------------------------------------
// WMMA GEMM, pixel-major: Y[b][p][o] = sum_c Xp[b][p][c] * W16[o][c] + bias[o]
// A = 16 pixels x 32 channels (contiguous per-lane loads)
// B = 32 channels x 16 out-channels (contiguous per-lane loads, W row-major)
// One wave: 16 pixels x 64 out-channels.  grid = (P/16, Cout/64, B), block 32.
// Writes optional f32 pixel-major and/or f16 pixel-major (scaled).
// ---------------------------------------------------------------------------
__global__ __launch_bounds__(32)
void psattn_gemm_kernel(const _Float16* __restrict__ Xp, const _Float16* __restrict__ W16,
                        const float* __restrict__ bias,
                        float* __restrict__ Yf32, _Float16* __restrict__ Yf16,
                        int Cin, int Cout, int P, float scale16) {
  const int lane = threadIdx.x & 31;
  const int m = lane & 15, hi = lane >> 4;
  const int p0 = blockIdx.x * 16;
  const int o0 = blockIdx.y * 64;
  const int b  = blockIdx.z;

  v8f acc[4];
#pragma unroll
  for (int t = 0; t < 4; ++t) acc[t] = vzero8();

  const _Float16* arow = Xp + ((size_t)(b * P + p0 + m)) * Cin;

  for (int c0 = 0; c0 < Cin; c0 += 32) {
    // A tile: rows = pixels p0..p0+15, K = c0..c0+31 (f16 A striping)
    v16h a;
    __builtin_prefetch(arow + c0 + 32, 0, 1);
#pragma unroll
    for (int e = 0; e < 8; ++e) a[e]     = arow[c0 + hi * 16 + e];
#pragma unroll
    for (int e = 0; e < 8; ++e) a[8 + e] = arow[c0 + hi * 16 + 8 + e];

#pragma unroll
    for (int t = 0; t < 4; ++t) {
      // B tile: B(k,n) = W16[(o0+t*16+n)*Cin + c0 + k]; lane = column n,
      // K = hi*16 + e -> 16 contiguous halfs
      v16h bm;
      const _Float16* wrow = W16 + (size_t)(o0 + t * 16 + m) * Cin + c0 + hi * 16;
#pragma unroll
      for (int e = 0; e < 16; ++e) bm[e] = wrow[e];
      acc[t] = wmma_f16(a, bm, acc[t]);
    }
  }

#pragma unroll
  for (int t = 0; t < 4; ++t) {
    float bn = bias[o0 + t * 16 + m];   // N = lane&15 in D layout
#pragma unroll
    for (int r = 0; r < 8; ++r) {
      int prow = p0 + r + 8 * hi;       // M = r + 8*hi in D layout
      size_t idx = ((size_t)(b * P + prow)) * Cout + o0 + t * 16 + m;
      float v = acc[t][r] + bn;
      if (Yf32) Yf32[idx] = v;
      if (Yf16) Yf16[idx] = (_Float16)(v * scale16);
    }
  }
}

// ---------------------------------------------------------------------------
// global_sim[b,h,m] = (1/N * sum_n q_scaled[n,:]) . k[m,:]  + hash-gumbel
// q16 pixel-major [b][p][256] (pre-scaled), kv16 pixel-major [b][pu][512]
// grid = B*NH, block = 256
// ---------------------------------------------------------------------------
__device__ __forceinline__ float psattn_hash_u(unsigned s) {
  s ^= s >> 17; s *= 0xed5ad4bbu; s ^= s >> 11; s *= 0xac4c1b51u;
  s ^= s >> 15; s *= 0x31848babu; s ^= s >> 14;
  return (float)(s >> 8) * (1.0f / 16777216.0f) + 1e-7f;
}

__global__ __launch_bounds__(256)
void psattn_gsim_kernel(const _Float16* __restrict__ q16, const _Float16* __restrict__ kv16,
                        float* __restrict__ noisy) {
  int bh = blockIdx.x;
  int b = bh >> 3, h = bh & 7;
  __shared__ float qbar[HD];
  __shared__ float part[256];
  int t = threadIdx.x;
  int d = t & 31, seg = t >> 5;
  const _Float16* qcol = q16 + (size_t)b * Npix * CDIM + h * HD + d;
  float s = 0.f;
  for (int n = seg; n < Npix; n += 8) s += (float)qcol[(size_t)n * CDIM];
  part[t] = s;
  __syncthreads();
  if (t < HD) {
    float a = 0.f;
#pragma unroll
    for (int k = 0; k < 8; ++k) a += part[k * 32 + t];
    qbar[t] = a * (1.0f / (float)Npix);
  }
  __syncthreads();
  const _Float16* kb = kv16 + (size_t)b * NuPix * 512 + h * 64;
  for (int mm = t; mm < NuPix; mm += 256) {
    float acc = 0.f;
#pragma unroll
    for (int d2 = 0; d2 < HD; ++d2) acc += qbar[d2] * (float)kb[(size_t)mm * 512 + d2];
    float u = psattn_hash_u((unsigned)(bh * 1024 + mm) * 0x9E3779B9u + 0x85ebca6bu);
    float g = -__logf(-__logf(u));
    noisy[(size_t)bh * NuPix + mm] = acc + g;
  }
}

// ---------------------------------------------------------------------------
// top-4 over noisy, expand to 16 fine indices, gather f_k/f_v rows (f16)
// fkv16 pixel-major [b][p][512]; tk/tv compact [bh][16][32]
// grid = B*NH, block = 32
// ---------------------------------------------------------------------------
__global__ __launch_bounds__(32)
void psattn_topk_gather_kernel(const float* __restrict__ noisy, const _Float16* __restrict__ fkv16,
                               _Float16* __restrict__ tk, _Float16* __restrict__ tv) {
  int bh = blockIdx.x;
  int b = bh >> 3, h = bh & 7;
  __shared__ int sidx[16];
  if (threadIdx.x == 0) {
    const float* nz = noisy + (size_t)bh * NuPix;
    int chosen[4];
    for (int ti = 0; ti < 4; ++ti) {
      float best = -1e30f; int bi = 0;
      for (int mz = 0; mz < NuPix; ++mz) {
        bool skip = false;
        for (int j = 0; j < ti; ++j) if (chosen[j] == mz) skip = true;
        float v = nz[mz];
        if (!skip && v > best) { best = v; bi = mz; }
      }
      chosen[ti] = bi;
    }
    for (int dh = 0; dh < 2; ++dh)
      for (int dw = 0; dw < 2; ++dw)
        for (int ti = 0; ti < 4; ++ti) {
          int mi = chosen[ti];
          int hy = (mi >> 5) * 2 + dh;   // Wu = 32, SCALE = 2
          int wx = (mi & 31) * 2 + dw;
          sidx[(dh * 2 + dw) * 4 + ti] = hy * 64 + wx;
        }
  }
  __syncthreads();
  const _Float16* fb = fkv16 + (size_t)b * Npix * 512 + h * 64;
  for (int i = threadIdx.x; i < 16 * HD; i += 32) {
    int j = i >> 5, d = i & 31;
    tk[(size_t)bh * 512 + i] = fb[(size_t)sidx[j] * 512 + d];
    tv[(size_t)bh * 512 + i] = fb[(size_t)sidx[j] * 512 + 32 + d];
  }
}

// ---------------------------------------------------------------------------
// Fused attention: coarse flash attention (1024 keys) + fine top-k attention
// + sigmoid gate fusion.  One wave owns 16 query rows of one (b,h).
// q16 [b][p][256] (head slice h*32), kv16 [b][pu][512] (k at h*64, v at +32)
// writes gated output into outacc f32 pixel-major [b][p][256]
// grid = (N/16, B*NH), block = 32
// ---------------------------------------------------------------------------
__global__ __launch_bounds__(32)
void psattn_attn_kernel(const _Float16* __restrict__ q16, const _Float16* __restrict__ kv16,
                        const _Float16* __restrict__ tk, const _Float16* __restrict__ tv,
                        const float* __restrict__ Wg, const float* __restrict__ bg,
                        float* __restrict__ outacc) {
  __shared__ _Float16 lds_p[16][32];
  __shared__ _Float16 lds_v[32][32];
  __shared__ float    lds_f[16][64];

  const int lane = threadIdx.x & 31;
  const int m = lane & 15, hi = lane >> 4;
  const int bh = blockIdx.y;
  const int b = bh >> 3, h = bh & 7;
  const int m0q = blockIdx.x * 16;

  // Q tile (16x32) in A striping; rows contiguous in pixel-major layout
  v16h qa;
  {
    const _Float16* qrow = q16 + ((size_t)(b * Npix + m0q + m)) * CDIM + h * HD;
#pragma unroll
    for (int e = 0; e < 8; ++e) qa[e]     = qrow[hi * 16 + e];
#pragma unroll
    for (int e = 0; e < 8; ++e) qa[8 + e] = qrow[hi * 16 + 8 + e];
  }

  float rm[8], rs[8];
#pragma unroll
  for (int r = 0; r < 8; ++r) { rm[r] = -INFINITY; rs[r] = 0.f; }
  v8f acc0 = vzero8(), acc1 = vzero8();

  const _Float16* kbase = kv16 + (size_t)b * NuPix * 512 + h * 64;

  for (int kk = 0; kk < NuPix; kk += 32) {
    // K B-tiles: B(d,n)=K[kk+n][d]; 16 contiguous halfs per lane
    v16h kb0, kb1;
    {
      const _Float16* r0 = kbase + (size_t)(kk + m) * 512 + hi * 16;
      const _Float16* r1 = kbase + (size_t)(kk + 16 + m) * 512 + hi * 16;
#pragma unroll
      for (int e = 0; e < 16; ++e) { kb0[e] = r0[e]; kb1[e] = r1[e]; }
    }
    v8f s0 = wmma_f16(qa, kb0, vzero8());
    v8f s1 = wmma_f16(qa, kb1, vzero8());

    // online softmax over this 32-key chunk
#pragma unroll
    for (int r = 0; r < 8; ++r) {
      float mx = fmaxf(s0[r], s1[r]);
#pragma unroll
      for (int off = 1; off < 16; off <<= 1) mx = fmaxf(mx, __shfl_xor(mx, off));
      float nm = fmaxf(rm[r], mx);
      float sc = __expf(rm[r] - nm);
      acc0[r] *= sc; acc1[r] *= sc; rs[r] *= sc;
      float p0 = __expf(s0[r] - nm);
      float p1 = __expf(s1[r] - nm);
      float ps = p0 + p1;
#pragma unroll
      for (int off = 1; off < 16; off <<= 1) ps += __shfl_xor(ps, off);
      rs[r] += ps; rm[r] = nm;
      lds_p[r + 8 * hi][m]      = (_Float16)p0;
      lds_p[r + 8 * hi][16 + m] = (_Float16)p1;
    }
    // stage V chunk (32 keys x 32 dims) in LDS, contiguous 64B per lane
    {
      const _Float16* vr = kbase + (size_t)(kk + lane) * 512 + 32;
#pragma unroll
      for (int e = 0; e < 32; ++e) lds_v[lane][e] = vr[e];
    }
    __syncthreads();

    // P (16x32) in A striping, V B-tiles for dims 0..15 / 16..31
    v16h pa, vb0, vb1;
#pragma unroll
    for (int e = 0; e < 8; ++e) pa[e]     = lds_p[m][hi * 16 + e];
#pragma unroll
    for (int e = 0; e < 8; ++e) pa[8 + e] = lds_p[m][hi * 16 + 8 + e];
#pragma unroll
    for (int e = 0; e < 16; ++e) { vb0[e] = lds_v[hi * 16 + e][m]; vb1[e] = lds_v[hi * 16 + e][16 + m]; }
    acc0 = wmma_f16(pa, vb0, acc0);
    acc1 = wmma_f16(pa, vb1, acc1);
    __syncthreads();
  }

  // finalize coarse
  float co0[8], co1[8];
#pragma unroll
  for (int r = 0; r < 8; ++r) { float inv = 1.f / rs[r]; co0[r] = acc0[r] * inv; co1[r] = acc1[r] * inv; }

  // ---- fine attention over the 16 gathered positions ----
  const _Float16* tkb = tk + (size_t)bh * 512;
  const _Float16* tvb = tv + (size_t)bh * 512;
  v16h fb;
#pragma unroll
  for (int e = 0; e < 16; ++e) fb[e] = tkb[(size_t)m * 32 + hi * 16 + e]; // B(d,j)=tk[j][d]
  v8f fs = wmma_f16(qa, fb, vzero8());

#pragma unroll
  for (int r = 0; r < 8; ++r) {
    float mx = fs[r];
#pragma unroll
    for (int off = 1; off < 16; off <<= 1) mx = fmaxf(mx, __shfl_xor(mx, off));
    float p = __expf(fs[r] - mx);
    float s = p;
#pragma unroll
    for (int off = 1; off < 16; off <<= 1) s += __shfl_xor(s, off);
    float fp = p / s;
    lds_p[r + 8 * hi][m]      = (_Float16)fp;
    lds_p[r + 8 * hi][16 + m] = (_Float16)0.f;   // pad K 16..31 with zeros
  }
  __syncthreads();

  v16h pa2, tb0, tb1;
#pragma unroll
  for (int e = 0; e < 8; ++e) pa2[e]     = lds_p[m][hi * 16 + e];
#pragma unroll
  for (int e = 0; e < 8; ++e) pa2[8 + e] = lds_p[m][hi * 16 + 8 + e];
#pragma unroll
  for (int e = 0; e < 16; ++e) {
    int j = hi * 16 + e;
    _Float16 a0 = (_Float16)0.f, a1 = (_Float16)0.f;
    if (j < 16) { a0 = tvb[(size_t)j * 32 + m]; a1 = tvb[(size_t)j * 32 + 16 + m]; }
    tb0[e] = a0; tb1[e] = a1;
  }
  v8f rf0 = wmma_f16(pa2, tb0, vzero8());
  v8f rf1 = wmma_f16(pa2, tb1, vzero8());
  __syncthreads();

  // ---- gate fusion: gate = sigmoid([coarse|refined] . Wg[d]) ----
#pragma unroll
  for (int r = 0; r < 8; ++r) {
    lds_f[r + 8 * hi][m]      = co0[r];
    lds_f[r + 8 * hi][16 + m] = co1[r];
    lds_f[r + 8 * hi][32 + m] = rf0[r];
    lds_f[r + 8 * hi][48 + m] = rf1[r];
  }
  __syncthreads();

  for (int t = 0; t < 16; ++t) {
    int idx = lane * 16 + t;
    int row = idx >> 5, d = idx & 31;
    float a = bg[d];
#pragma unroll 8
    for (int c = 0; c < 64; ++c) a += lds_f[row][c] * Wg[d * 64 + c];
    float g = 1.f / (1.f + __expf(-a));
    float ref = lds_f[row][32 + d];
    float coa = lds_f[row][d];
    outacc[((size_t)(b * Npix + m0q + row)) * CDIM + h * HD + d] = g * ref + (1.f - g) * coa;
  }
}

// ---------------------------------------------------------------------------
// Depthwise 7x7 conv on v (32x32) -> pe_small planar f32 [b][c][1024]
// v channel c lives at kv16 pixel-major column (c/32)*64 + 32 + c%32
// grid = B*256, block = 256
// ---------------------------------------------------------------------------
__global__ __launch_bounds__(256)
void psattn_peconv_kernel(const _Float16* __restrict__ kv16, const float* __restrict__ Wpe,
                          const float* __restrict__ bpe, float* __restrict__ pe_small) {
  int bc = blockIdx.x;
  int b = bc >> 8, c = bc & 255;
  int vo = (c >> 5) * 64 + 32 + (c & 31);
  const _Float16* vcol = kv16 + (size_t)b * NuPix * 512 + vo;
  const float* wk = Wpe + (size_t)c * 49;
  float bias = bpe[c];
  for (int pix = threadIdx.x; pix < NuPix; pix += 256) {
    int y = pix >> 5, x = pix & 31;
    float acc = bias;
#pragma unroll
    for (int ky = 0; ky < 7; ++ky) {
      int yy = y + ky - 3;
      if ((unsigned)yy >= 32u) continue;
#pragma unroll
      for (int kx = 0; kx < 7; ++kx) {
        int xx = x + kx - 3;
        if ((unsigned)xx >= 32u) continue;
        acc += wk[ky * 7 + kx] * (float)vcol[(size_t)(yy * 32 + xx) * 512];
      }
    }
    pe_small[((size_t)b * CDIM + c) * NuPix + pix] = acc;
  }
}

// ---------------------------------------------------------------------------
// Bilinear 2x upsample (half-pixel, edge clamp), add into outacc pixel-major
// i enumerates [b][p][c], c fastest
// ---------------------------------------------------------------------------
__global__ void psattn_upsample_add_kernel(const float* __restrict__ pe_small,
                                           float* __restrict__ outacc, size_t total) {
  size_t i = (size_t)blockIdx.x * blockDim.x + threadIdx.x;
  if (i >= total) return;
  int c = (int)(i & 255);
  int p = (int)((i >> 8) & 4095);
  int b = (int)(i >> 20);
  int Y = p >> 6, X = p & 63;
  float fy = fminf(fmaxf((Y + 0.5f) * 0.5f - 0.5f, 0.f), 31.f);
  float fx = fminf(fmaxf((X + 0.5f) * 0.5f - 0.5f, 0.f), 31.f);
  int y0 = (int)fy, x0 = (int)fx;
  int y1 = min(y0 + 1, 31), x1 = min(x0 + 1, 31);
  float wy = fy - y0, wx = fx - x0;
  const float* s = pe_small + ((size_t)b * CDIM + c) * NuPix;
  float v = (1.f - wy) * ((1.f - wx) * s[y0 * 32 + x0] + wx * s[y0 * 32 + x1]) +
            wy * ((1.f - wx) * s[y1 * 32 + x0] + wx * s[y1 * 32 + x1]);
  outacc[i] += v;
}

// ---------------------------------------------------------------------------
// Transpose pixel-major f32 [b][p][256] -> planar f32 [b][c][p] (d_out NCHW)
// ---------------------------------------------------------------------------
__global__ void psattn_trout_kernel(const float* __restrict__ srcpm, float* __restrict__ dst,
                                    size_t total) {
  size_t i = (size_t)blockIdx.x * blockDim.x + threadIdx.x;
  if (i >= total) return;
  int p = (int)(i & 4095);
  int c = (int)((i >> 12) & 255);
  int b = (int)(i >> 20);
  dst[i] = srcpm[((size_t)(b * Npix + p)) * CDIM + c];
}

// ---------------------------------------------------------------------------
extern "C" void kernel_launch(void* const* d_in, const int* in_sizes, int n_in,
                              void* d_out, int out_size, void* d_ws, size_t ws_size,
                              hipStream_t stream) {
  const float* x     = (const float*)d_in[0];
  const float* upper = (const float*)d_in[1];
  const float* Wq    = (const float*)d_in[2];
  const float* bq    = (const float*)d_in[3];
  const float* Wkv   = (const float*)d_in[4];
  const float* bkv   = (const float*)d_in[5];
  const float* Wproj = (const float*)d_in[6];
  const float* bproj = (const float*)d_in[7];
  const float* Wpe   = (const float*)d_in[8];
  const float* bpe   = (const float*)d_in[9];
  const float* Wg    = (const float*)d_in[10];
  const float* bg    = (const float*)d_in[11];
  float* out = (float*)d_out;

  char* base = (char*)d_ws;
  size_t off = 0;
  auto carve = [&](size_t bytes) { char* p = base + off; off = (off + bytes + 255) & ~(size_t)255; return p; };

  _Float16* xp16    = (_Float16*)carve((size_t)Bn * Npix  * CDIM * 2);
  _Float16* up16    = (_Float16*)carve((size_t)Bn * NuPix * CDIM * 2);
  _Float16* wq16    = (_Float16*)carve((size_t)CDIM * CDIM * 2);
  _Float16* wkv16   = (_Float16*)carve((size_t)512  * CDIM * 2);
  _Float16* wproj16 = (_Float16*)carve((size_t)CDIM * CDIM * 2);
  _Float16* q16     = (_Float16*)carve((size_t)Bn * Npix  * CDIM * 2);
  _Float16* kv16    = (_Float16*)carve((size_t)Bn * NuPix * 512  * 2);
  _Float16* fkv16   = (_Float16*)carve((size_t)Bn * Npix  * 512  * 2);
  float*    outacc  = (float*)   carve((size_t)Bn * Npix  * CDIM * 4);
  _Float16* outp16  = (_Float16*)carve((size_t)Bn * Npix  * CDIM * 2);
  float*    projpm  = (float*)   carve((size_t)Bn * Npix  * CDIM * 4);
  float*    pe_sm   = (float*)   carve((size_t)Bn * CDIM  * NuPix * 4);
  float*    noisy   = (float*)   carve((size_t)Bn * NH * NuPix * 4);
  _Float16* tk      = (_Float16*)carve((size_t)Bn * NH * 16 * HD * 2);
  _Float16* tv      = (_Float16*)carve((size_t)Bn * NH * 16 * HD * 2);
  (void)ws_size; (void)in_sizes; (void)n_in; (void)out_size;

  // 1) pack inputs to f16 pixel-major, weights to f16
  {
    size_t t1 = (size_t)Bn * CDIM * Npix;
    psattn_trpack_kernel<<<(unsigned)((t1 + 255) / 256), 256, 0, stream>>>(x, xp16, CDIM, Npix, t1);
    size_t t2 = (size_t)Bn * CDIM * NuPix;
    psattn_trpack_kernel<<<(unsigned)((t2 + 255) / 256), 256, 0, stream>>>(upper, up16, CDIM, NuPix, t2);
    size_t w1 = (size_t)CDIM * CDIM;
    psattn_cvt16_kernel<<<(unsigned)((w1 + 255) / 256), 256, 0, stream>>>(Wq, wq16, 1.0f, w1);
    size_t w2 = (size_t)512 * CDIM;
    psattn_cvt16_kernel<<<(unsigned)((w2 + 255) / 256), 256, 0, stream>>>(Wkv, wkv16, 1.0f, w2);
    psattn_cvt16_kernel<<<(unsigned)((w1 + 255) / 256), 256, 0, stream>>>(Wproj, wproj16, 1.0f, w1);
  }

  // 2) conv1x1 GEMMs (WMMA); q pre-scaled by hd^-0.5, outputs in attention layout
  psattn_gemm_kernel<<<dim3(Npix / 16, CDIM / 64, Bn), 32, 0, stream>>>(
      xp16, wq16, bq, nullptr, q16, CDIM, CDIM, Npix, SCALE_QK);
  psattn_gemm_kernel<<<dim3(NuPix / 16, 512 / 64, Bn), 32, 0, stream>>>(
      up16, wkv16, bkv, nullptr, kv16, CDIM, 512, NuPix, 1.0f);
  psattn_gemm_kernel<<<dim3(Npix / 16, 512 / 64, Bn), 32, 0, stream>>>(
      xp16, wkv16, bkv, nullptr, fkv16, CDIM, 512, Npix, 1.0f);

  // 3) global similarity (+gumbel) and top-k gather
  psattn_gsim_kernel<<<Bn * NH, 256, 0, stream>>>(q16, kv16, noisy);
  psattn_topk_gather_kernel<<<Bn * NH, 32, 0, stream>>>(noisy, fkv16, tk, tv);

  // 4) fused coarse(flash) + fine attention + gate -> outacc (pixel-major f32)
  psattn_attn_kernel<<<dim3(Npix / 16, Bn * NH), 32, 0, stream>>>(q16, kv16, tk, tv, Wg, bg, outacc);

  // 5) depthwise PE conv + bilinear upsample-add into outacc
  psattn_peconv_kernel<<<Bn * CDIM, 256, 0, stream>>>(kv16, Wpe, bpe, pe_sm);
  {
    size_t t = (size_t)Bn * Npix * CDIM;
    psattn_upsample_add_kernel<<<(unsigned)((t + 255) / 256), 256, 0, stream>>>(pe_sm, outacc, t);
  }

  // 6) final projection (WMMA GEMM) -> planar d_out
  {
    size_t t = (size_t)Bn * Npix * CDIM;
    psattn_cvt16_kernel<<<(unsigned)((t + 255) / 256), 256, 0, stream>>>(outacc, outp16, 1.0f, t);
    psattn_gemm_kernel<<<dim3(Npix / 16, CDIM / 64, Bn), 32, 0, stream>>>(
        outp16, wproj16, bproj, projpm, nullptr, CDIM, CDIM, Npix, 1.0f);
    psattn_trout_kernel<<<(unsigned)((t + 255) / 256), 256, 0, stream>>>(projpm, out, t);
  }
}
